// MultiHeadGraphAttention_70179765616702
// MI455X (gfx1250) — matrix operand order repaired and checked by reading, not compiled, not run on hardware
//
#include <hip/hip_runtime.h>
#include <stdint.h>

#define N_HEAD 4
#define BS 8192
#define F_IN 64
#define F_OUT 64
#define LOG2E 1.4426950408889634f

typedef __attribute__((ext_vector_type(16))) _Float16 v16h;
typedef __attribute__((ext_vector_type(8)))  float    v8f;
typedef int v4i_vs __attribute__((vector_size(16)));   // matches async builtin param type

#if defined(__gfx1250__) && __has_builtin(__builtin_amdgcn_global_load_async_to_lds_b128)
#define HAVE_ASYNC 1
#else
#define HAVE_ASYNC 0
#endif

__device__ __forceinline__ void wait_async0() {
#if HAVE_ASYNC
#if __has_builtin(__builtin_amdgcn_s_wait_asynccnt)
  __builtin_amdgcn_s_wait_asynccnt(0);
#else
  asm volatile("s_wait_asynccnt 0" ::: "memory");
#endif
#endif
}

// ---------------------------------------------------------------------------
// Phase 1: h_prime = h @ W[head] (fp32); s = hp.a_src, d = hp.a_dst;
// also dpos = exp2(d*log2e), dneg = exp2(0.2*d*log2e)  (factored-exp tables).
// Store V = (f16)h_prime pre-swizzled into the WMMA B-operand layout:
//   idx = head*524288 + kb*2048 + n*512 + (gh*16 + l16)*16 + t
// for row j: kb=j>>5, gh=(j>>3)&1, t=(j&7)|(((j>>4)&1)<<3); for col o: n=o>>4, l16=o&15.
// ---------------------------------------------------------------------------
__global__ __launch_bounds__(256) void gat_project_kernel(
    const float* __restrict__ h, const float* __restrict__ w,
    const float* __restrict__ a_src, const float* __restrict__ a_dst,
    _Float16* __restrict__ Vws, float* __restrict__ s_arr, float* __restrict__ d_arr,
    float* __restrict__ dpos, float* __restrict__ dneg)
{
  const int head = blockIdx.y;
  const int b = (int)blockIdx.x * 256 + (int)threadIdx.x;

  const float4* __restrict__ h4 = (const float4*)(h + (size_t)b * F_IN);
  const float* __restrict__ wh = w + head * (F_IN * F_OUT);
  const float* __restrict__ as = a_src + head * F_OUT;
  const float* __restrict__ ad = a_dst + head * F_OUT;

  const int kb = b >> 5;
  const int t  = (b & 7) | (((b >> 4) & 1) << 3);
  const int gh = (b >> 3) & 1;
  _Float16* __restrict__ vdst =
      Vws + (size_t)head * (BS * F_OUT) + (size_t)kb * 2048 + gh * 256 + t;

  float s = 0.0f, dd = 0.0f;
  for (int oc = 0; oc < 4; ++oc) {
    float accv[16];
#pragma unroll
    for (int oo = 0; oo < 16; ++oo) accv[oo] = 0.0f;
    for (int i4 = 0; i4 < 16; ++i4) {
      const float4 hv = h4[i4];
      const float* w0 = wh + (i4 * 4) * F_OUT + oc * 16;  // wave-uniform -> SMEM
#pragma unroll
      for (int oo = 0; oo < 16; ++oo) accv[oo] = fmaf(hv.x, w0[oo], accv[oo]);
#pragma unroll
      for (int oo = 0; oo < 16; ++oo) accv[oo] = fmaf(hv.y, w0[F_OUT + oo], accv[oo]);
#pragma unroll
      for (int oo = 0; oo < 16; ++oo) accv[oo] = fmaf(hv.z, w0[2 * F_OUT + oo], accv[oo]);
#pragma unroll
      for (int oo = 0; oo < 16; ++oo) accv[oo] = fmaf(hv.w, w0[3 * F_OUT + oo], accv[oo]);
    }
#pragma unroll
    for (int oo = 0; oo < 16; ++oo) {
      const int o = oc * 16 + oo;
      const float hp = accv[oo];
      s  = fmaf(hp, as[o], s);
      dd = fmaf(hp, ad[o], dd);
      vdst[oc * 512 + oo * 16] = (_Float16)hp;
    }
  }
  s_arr[head * BS + b] = s;
  d_arr[head * BS + b] = dd;
  dpos[head * BS + b] = __builtin_amdgcn_exp2f(dd * LOG2E);
  dneg[head * BS + b] = __builtin_amdgcn_exp2f(dd * (0.2f * LOG2E));
}

// ---------------------------------------------------------------------------
// Phase 1.5: per-head C = leakyrelu(max_i s_i + max_j d_j) * log2(e)
// (row-uniform shift -> softmax invariant; keeps P <= 1 for f16)
// ---------------------------------------------------------------------------
__global__ __launch_bounds__(256) void gat_reduce_kernel(
    const float* __restrict__ s_arr, const float* __restrict__ d_arr,
    float* __restrict__ cvec)
{
  __shared__ float red[256];
  const int head = blockIdx.x;
  const int tid = (int)threadIdx.x;
  float smax = -1e30f, dmax = -1e30f;
  for (int i = tid; i < BS; i += 256) {
    smax = fmaxf(smax, s_arr[head * BS + i]);
    dmax = fmaxf(dmax, d_arr[head * BS + i]);
  }
  red[tid] = smax;
  __syncthreads();
  for (int s = 128; s > 0; s >>= 1) {
    if (tid < s) red[tid] = fmaxf(red[tid], red[tid + s]);
    __syncthreads();
  }
  const float sm = red[0];
  __syncthreads();
  red[tid] = dmax;
  __syncthreads();
  for (int s = 128; s > 0; s >>= 1) {
    if (tid < s) red[tid] = fmaxf(red[tid], red[tid + s]);
    __syncthreads();
  }
  if (tid == 0) {
    const float x = sm + red[0];
    const float y = fmaxf(x, 0.2f * x);
    cvec[head] = y * LOG2E;
  }
}

// ---------------------------------------------------------------------------
// Phase 2: flash-style P@V per head with v_wmma_f32_16x16x32_f16.
// Block = 8 waves; wave owns 16 rows x 64 cols; V double-buffered in LDS via
// async global->LDS copies. P tile built with NO transcendentals:
//   P_ij = max(A_i*Bp_j, C_i*Bn_j),  A/C per-lane hoisted, Bp/Bn streamed.
// Row sums Z computed by a 5th WMMA against an all-ones B tile (same C-layout
// row mapping as the output accumulators).
// A-tile K order: K(t) = 16*(t/8) + 8*(lane/16) + (t%8) -> contiguous float4s.
// ---------------------------------------------------------------------------
__global__ __launch_bounds__(256) void gat_attn_kernel(
    const _Float16* __restrict__ Vws,
    const float* __restrict__ s_arr,
    const float* __restrict__ dpos,
    const float* __restrict__ dneg,
    const float* __restrict__ cvec,
    const float* __restrict__ bias,
    float* __restrict__ out)
{
  __shared__ __align__(32) _Float16 Vs[2][8192];  // 2 x 16 KB

  const int head = blockIdx.y;
  const int tid  = (int)threadIdx.x;
  const int wave = tid >> 5;
  const int lane = tid & 31;
  const int g    = lane >> 4;
  const int l16  = lane & 15;
  const int i0   = (int)blockIdx.x * 128 + wave * 16;

  const float sM = s_arr[head * BS + i0 + l16];
  const float C2 = cvec[head];
  // exp factors for this lane's P row (exact: exp/max commute)
  const float Ai = __builtin_amdgcn_exp2f(fmaf(sM, LOG2E, -C2));
  const float Ci = __builtin_amdgcn_exp2f(fmaf(sM, 0.2f * LOG2E, -C2));

  const float* __restrict__ dp = dpos + head * BS;
  const float* __restrict__ dn = dneg + head * BS;
  const _Float16* __restrict__ vg = Vws + (size_t)head * (BS * F_OUT);

  v16h vones;
#pragma unroll
  for (int t = 0; t < 16; ++t) vones[t] = (_Float16)1.0f;

  v8f acc0 = {}, acc1 = {}, acc2 = {}, acc3 = {}, accz = {};

  auto stage = [&](int kc, int buf) {
    const _Float16* gsrc = vg + (size_t)kc * 8192 + tid * 32;  // 64 B per thread
    _Float16* ldst = &Vs[buf][tid * 32];
#if HAVE_ASYNC
#pragma unroll
    for (int q = 0; q < 4; ++q) {
      __builtin_amdgcn_global_load_async_to_lds_b128(
          (v4i_vs*)(gsrc + q * 8), (v4i_vs*)(ldst + q * 8), 0, 0);
    }
#else
#pragma unroll
    for (int q = 0; q < 4; ++q)
      ((float4*)ldst)[q] = ((const float4*)gsrc)[q];
#endif
  };

  stage(0, 0);

  for (int kc = 0; kc < 64; ++kc) {
    const int buf = kc & 1;
    wait_async0();
    __syncthreads();  // chunk kc visible to all waves; prev compute done by all
    if (kc + 1 < 64) stage(kc + 1, buf ^ 1);

#pragma unroll
    for (int kbl = 0; kbl < 4; ++kbl) {
      const int j0 = kc * 128 + kbl * 32;
      const float4 p0 = *(const float4*)(dp + j0 + 8 * g);
      const float4 p1 = *(const float4*)(dp + j0 + 8 * g + 4);
      const float4 p2 = *(const float4*)(dp + j0 + 16 + 8 * g);
      const float4 p3 = *(const float4*)(dp + j0 + 16 + 8 * g + 4);
      const float4 n0 = *(const float4*)(dn + j0 + 8 * g);
      const float4 n1 = *(const float4*)(dn + j0 + 8 * g + 4);
      const float4 n2 = *(const float4*)(dn + j0 + 16 + 8 * g);
      const float4 n3 = *(const float4*)(dn + j0 + 16 + 8 * g + 4);
      const float bp[16] = {p0.x, p0.y, p0.z, p0.w, p1.x, p1.y, p1.z, p1.w,
                            p2.x, p2.y, p2.z, p2.w, p3.x, p3.y, p3.z, p3.w};
      const float bn[16] = {n0.x, n0.y, n0.z, n0.w, n1.x, n1.y, n1.z, n1.w,
                            n2.x, n2.y, n2.z, n2.w, n3.x, n3.y, n3.z, n3.w};
      v16h a;
#pragma unroll
      for (int t = 0; t < 16; ++t) {
        // P = exp2(leakyrelu(s+d)*log2e - C) == max(Ai*Bp, Ci*Bn)
        a[t] = (_Float16)fmaxf(Ai * bp[t], Ci * bn[t]);
      }
      const _Float16* bbase = &Vs[buf][kbl * 2048 + lane * 16];
      const v16h b0 = *(const v16h*)(bbase + 0);
      const v16h b1 = *(const v16h*)(bbase + 512);
      const v16h b2 = *(const v16h*)(bbase + 1024);
      const v16h b3 = *(const v16h*)(bbase + 1536);
      acc0 = __builtin_amdgcn_wmma_f32_16x16x32_f16(false, a, false, b0, (short)0, acc0, false, false);
      acc1 = __builtin_amdgcn_wmma_f32_16x16x32_f16(false, a, false, b1, (short)0, acc1, false, false);
      acc2 = __builtin_amdgcn_wmma_f32_16x16x32_f16(false, a, false, b2, (short)0, acc2, false, false);
      acc3 = __builtin_amdgcn_wmma_f32_16x16x32_f16(false, a, false, b3, (short)0, acc3, false, false);
      accz = __builtin_amdgcn_wmma_f32_16x16x32_f16(false, a, false, vones, (short)0, accz, false, false);
    }
  }

  const float bv0 = bias[0 + l16], bv1 = bias[16 + l16],
              bv2 = bias[32 + l16], bv3 = bias[48 + l16];
#pragma unroll
  for (int r = 0; r < 8; ++r) {
    const float inv = 1.0f / accz[r];        // Z for row M = r + 8*(lane/16)
    const int row = i0 + r + 8 * g;
    float* orow = out + (size_t)row * (N_HEAD * F_OUT) + head * F_OUT + l16;
    orow[0]  = acc0[r] * inv + bv0;
    orow[16] = acc1[r] * inv + bv1;
    orow[32] = acc2[r] * inv + bv2;
    orow[48] = acc3[r] * inv + bv3;
  }
}

// ---------------------------------------------------------------------------
extern "C" void kernel_launch(void* const* d_in, const int* in_sizes, int n_in,
                              void* d_out, int out_size, void* d_ws, size_t ws_size,
                              hipStream_t stream) {
  const float* h     = (const float*)d_in[0];
  const float* w     = (const float*)d_in[1];
  const float* a_src = (const float*)d_in[2];
  const float* a_dst = (const float*)d_in[3];
  const float* bias  = (const float*)d_in[4];
  float* out = (float*)d_out;

  // Workspace: [0,4MB) V f16 swizzled; s, d, dpos, dneg (128KB each); C (16B)
  char* ws = (char*)d_ws;
  _Float16* Vws  = (_Float16*)ws;
  float*    sarr = (float*)(ws + (4u << 20));
  float*    darr = (float*)(ws + (4u << 20) + (128u << 10));
  float*    dpos = (float*)(ws + (4u << 20) + (256u << 10));
  float*    dneg = (float*)(ws + (4u << 20) + (384u << 10));
  float*    cvec = (float*)(ws + (4u << 20) + (512u << 10));

  gat_project_kernel<<<dim3(BS / 256, N_HEAD), 256, 0, stream>>>(
      h, w, a_src, a_dst, Vws, sarr, darr, dpos, dneg);
  gat_reduce_kernel<<<dim3(N_HEAD), 256, 0, stream>>>(sarr, darr, cvec);
  gat_attn_kernel<<<dim3(BS / 128, N_HEAD), 256, 0, stream>>>(
      Vws, sarr, dpos, dneg, cvec, bias, out);
}
